// SpatialMultiHeadAttention_57775900066185
// MI455X (gfx1250) — matrix operand order repaired and checked by reading, not compiled, not run on hardware
//
#include <hip/hip_runtime.h>
#include <hip/hip_bf16.h>

#define NH 8
#define HD 32
#define ED 256
#define BB 2
#define NN 2048
#define BN (BB * NN) /* 4096 rows */

#define SROW 2056            /* LDS score-row stride in floats (2048 + 8) */
#define S_FLOATS (16 * SROW) /* 32896 */
#define P_FLOATS (2 * 4 * 8 * 32) /* 2048 partial-accum floats */
#define SMEM_FLOATS (S_FLOATS + P_FLOATS + 16)
#define SMEM_BYTES (SMEM_FLOATS * sizeof(float)) /* ~139.7 KB < 320 KB WGP LDS */

typedef __bf16 v16bf __attribute__((ext_vector_type(16)));
typedef __bf16 v2bf  __attribute__((ext_vector_type(2)));
typedef float  v8f   __attribute__((ext_vector_type(8)));

// ---------- conversion helpers ----------

__device__ __forceinline__ v2bf cvt_pk(float a, float b) {
#if __has_builtin(__builtin_amdgcn_cvt_pk_bf16_f32)
    return __builtin_amdgcn_cvt_pk_bf16_f32(a, b);   // v_cvt_pk_bf16_f32
#else
    v2bf r; r[0] = (__bf16)a; r[1] = (__bf16)b; return r;
#endif
}

union FragU {
    v16bf v;
    v2bf  p[8];
    uint4 q[2];
};

// A-matrix 16x32 bf16 fragment: lane half=0 -> K 0..15,
// half=1 -> K 16..31 (matches B).  `p` points at row start + k0 (contiguous K).
__device__ __forceinline__ v16bf load_a_f32(const float* p, int half) {
    const float4* p0 = (const float4*)(p + half * 16);
    const float4* p1 = (const float4*)(p + half * 16 + 8);
    float4 a0 = p0[0], a1 = p0[1], b0 = p1[0], b1 = p1[1];
    FragU u;
    u.p[0] = cvt_pk(a0.x, a0.y); u.p[1] = cvt_pk(a0.z, a0.w);
    u.p[2] = cvt_pk(a1.x, a1.y); u.p[3] = cvt_pk(a1.z, a1.w);
    u.p[4] = cvt_pk(b0.x, b0.y); u.p[5] = cvt_pk(b0.z, b0.w);
    u.p[6] = cvt_pk(b1.x, b1.y); u.p[7] = cvt_pk(b1.z, b1.w);
    return u.v;
}
__device__ __forceinline__ v16bf load_a_bf16(const __bf16* __restrict__ p, int half) {
    FragU u;
    u.q[0] = *(const uint4*)(p + half * 16);
    u.q[1] = *(const uint4*)(p + half * 16 + 8);
    return u.v;
}

// B-matrix 32x16 bf16 fragment: lanes 0-15 hold K=0..15, lanes 16-31 hold
// K=16..31 (2 K per VGPR).  `p` points at the Bᵀ row (contiguous K).
__device__ __forceinline__ v16bf load_b_f32(const float* __restrict__ p, int half) {
    const float4* q = (const float4*)(p + half * 16);
    float4 a = q[0], b = q[1], c = q[2], d = q[3];
    FragU u;
    u.p[0] = cvt_pk(a.x, a.y); u.p[1] = cvt_pk(a.z, a.w);
    u.p[2] = cvt_pk(b.x, b.y); u.p[3] = cvt_pk(b.z, b.w);
    u.p[4] = cvt_pk(c.x, c.y); u.p[5] = cvt_pk(c.z, c.w);
    u.p[6] = cvt_pk(d.x, d.y); u.p[7] = cvt_pk(d.z, d.w);
    return u.v;
}
__device__ __forceinline__ v16bf load_b_bf16(const __bf16* __restrict__ p, int half) {
    const uint4* q = (const uint4*)(p + half * 16);
    FragU u;
    u.q[0] = q[0]; u.q[1] = q[1];
    return u.v;
}

__device__ __forceinline__ v8f wmma_bf16(v16bf a, v16bf b, v8f c) {
    return __builtin_amdgcn_wmma_f32_16x16x32_bf16(false, a, false, b, (short)0, c,
                                                   false, false);
}

// ---------- kernel 1: Q/K/V projections (X @ Wᵀ + b), bf16 outputs ----------
__global__ __launch_bounds__(32)
void proj_qkv_kernel(const float* __restrict__ X,
                     const float* __restrict__ Wq, const float* __restrict__ bq,
                     const float* __restrict__ Wk, const float* __restrict__ bk,
                     const float* __restrict__ Wv, const float* __restrict__ bv,
                     __bf16* __restrict__ Qb, __bf16* __restrict__ Kb,
                     __bf16* __restrict__ Vt) {
    const int which = blockIdx.z;                // 0=Q 1=K 2=V
    const float* W    = (which == 0) ? Wq : (which == 1) ? Wk : Wv;
    const float* bias = (which == 0) ? bq : (which == 1) ? bk : bv;
    const int rt = blockIdx.x;                   // 0..255 row tiles of 16
    const int ct = blockIdx.y;                   // 0..15  col tiles of 16
    const int lane = threadIdx.x, half = lane >> 4, l = lane & 15;

    const float* arow = X + (size_t)(rt * 16 + l) * ED;
    const float* brow = W + (size_t)(ct * 16 + l) * ED;
    v8f acc = {};
#pragma unroll
    for (int k0 = 0; k0 < ED; k0 += 32) {
        v16bf a = load_a_f32(arow + k0, half);
        v16bf b = load_b_f32(brow + k0, half);
        acc = wmma_bf16(a, b, acc);
    }
    const float qscale = 0.17677669529663688f;   // 1/sqrt(32)
#pragma unroll
    for (int i = 0; i < 8; ++i) {
        int row = rt * 16 + half * 8 + i;        // 0..4095 (= b*N + n)
        int col = ct * 16 + l;                   // 0..255  (= h*Hd + d)
        float v = acc[i] + bias[col];
        int b_ = row >> 11, n = row & (NN - 1);
        int h  = col >> 5,  d = col & (HD - 1);
        if (which == 0)
            Qb[(((size_t)(b_ * NH + h) * NN + n) * HD) + d] = (__bf16)(v * qscale);
        else if (which == 1)
            Kb[(((size_t)(b_ * NH + h) * NN + n) * HD) + d] = (__bf16)v;
        else
            Vt[(((size_t)(b_ * NH + h) * HD + d) * NN) + n] = (__bf16)v;
    }
}

// ---------- kernel 2 (fused): scores+bias+mask -> softmax -> attn out + attn@V
// One block (8 waves) owns a 16-row strip of one (b,h): the full 16x2048 score
// strip lives in LDS, softmax is done in LDS, normalized attn is written to
// global once, and context (attn@V) is computed straight out of LDS with the
// 1/rowsum folded into the final scale.
__global__ __launch_bounds__(256)
void fused_attn_kernel(const __bf16* __restrict__ Qb, const __bf16* __restrict__ Kb,
                       const __bf16* __restrict__ Vt,
                       const float* __restrict__ positions,
                       const unsigned char* __restrict__ mask,
                       const float* __restrict__ log_alpha,
                       const float* __restrict__ beta_,
                       float* __restrict__ attn, __bf16* __restrict__ ctx) {
    extern __shared__ float smem[];
    float* S      = smem;                 // [16][SROW] scores, then exp values
    float* P      = smem + S_FLOATS;      // [2][4][8][32] context partials
    float* invRow = P + P_FLOATS;         // [16] 1/rowsum

    const int bh = blockIdx.y, b = bh >> 3, h = bh & 7;
    const int rt = blockIdx.x;            // 0..127 row strips of 16
    const int w    = threadIdx.x >> 5;    // wave 0..7
    const int lane = threadIdx.x & 31;
    const int half = lane >> 4, l = lane & 15;

    const __bf16* Qp = Qb + (size_t)bh * NN * HD;
    const __bf16* Kp = Kb + (size_t)bh * NN * HD;

    float la = log_alpha[h];
    float alpha = (la > 20.f) ? la : log1pf(__expf(la));   // softplus
    float bet = beta_[h];

    // ---- phase 1: scores + bias + mask into LDS (wave w: 16 column tiles) ----
    v16bf aq = load_a_bf16(Qp + (size_t)(rt * 16 + l) * HD, half);

    float prx[8], pry[8], prz[8];
    int mrow[8];
#pragma unroll
    for (int i = 0; i < 8; ++i) {
        int row = rt * 16 + half * 8 + i;
        const float* pp = positions + ((size_t)b * NN + row) * 3;
        prx[i] = pp[0]; pry[i] = pp[1]; prz[i] = pp[2];
        mrow[i] = mask[(size_t)b * NN + row];
    }

    for (int t = 0; t < 16; ++t) {
        int j = (w * 16 + t) * 16 + l;             // this lane's column
        v16bf bb = load_b_bf16(Kp + (size_t)j * HD, half);
        v8f c = {};
        c = wmma_bf16(aq, bb, c);
        const float* pj = positions + ((size_t)b * NN + j) * 3;
        float pjx = pj[0], pjy = pj[1], pjz = pj[2];
        int mj = mask[(size_t)b * NN + j];
#pragma unroll
        for (int i = 0; i < 8; ++i) {
            float dx = prx[i] - pjx, dy = pry[i] - pjy, dz = prz[i] - pjz;
            float s = c[i] - alpha * (dx * dx + dy * dy + dz * dz) + bet;
            s = (mrow[i] && mj) ? s : -__builtin_huge_valf();
            S[(half * 8 + i) * SROW + j] = s;
        }
    }
    __syncthreads();

    // ---- phase 2: softmax in LDS; wave w owns rows 2w (lanes 0-15) / 2w+1 ----
    {
        const int row = 2 * w + half;              // 0..15
        const int g = l;                           // lane-in-group 0..15
        float* Sr = S + row * SROW;
        float m = -__builtin_huge_valf();
#pragma unroll 4
        for (int k = 0; k < 32; ++k) {
            float4 v = *(float4*)(Sr + 4 * g + 64 * k);
            m = fmaxf(m, fmaxf(fmaxf(v.x, v.y), fmaxf(v.z, v.w)));
        }
#pragma unroll
        for (int off = 8; off; off >>= 1) m = fmaxf(m, __shfl_xor(m, off, 16));
        float sum = 0.f;
#pragma unroll 4
        for (int k = 0; k < 32; ++k) {
            float4 v = *(float4*)(Sr + 4 * g + 64 * k);
            v.x = __expf(v.x - m); v.y = __expf(v.y - m);
            v.z = __expf(v.z - m); v.w = __expf(v.w - m);
            *(float4*)(Sr + 4 * g + 64 * k) = v;   // keep UNNORMALIZED exp
            sum += v.x + v.y + v.z + v.w;
        }
#pragma unroll
        for (int off = 8; off; off >>= 1) sum += __shfl_xor(sum, off, 16);
        float inv = 1.0f / sum;
        if (g == 0) invRow[row] = inv;
        // single normalized write of this attn row to global
        float* gRow = attn + ((size_t)bh * NN + rt * 16 + row) * NN;
#pragma unroll 4
        for (int k = 0; k < 32; ++k) {
            float4 v = *(float4*)(Sr + 4 * g + 64 * k);
            v.x *= inv; v.y *= inv; v.z *= inv; v.w *= inv;
            *(float4*)(gRow + 4 * g + 64 * k) = v;
        }
    }
    __syncthreads();

    // ---- phase 3: context = exp(S) @ V from LDS; wave w -> d-tile w>>2,
    //      K-quarter (w&3)*512; 1/rowsum folded into the final scale ----
    {
        const int t  = w >> 2;                     // 0..1
        const int kq = (w & 3) * 512;
        const __bf16* Bp = Vt + (size_t)bh * HD * NN + (size_t)(t * 16 + l) * NN;
        v8f acc = {};
        for (int k0 = kq; k0 < kq + 512; k0 += 32) {
            v16bf a  = load_a_f32(S + (size_t)l * SROW + k0, half);  // LDS reads
            v16bf bb = load_b_bf16(Bp + k0, half);
            acc = wmma_bf16(a, bb, acc);
        }
        float* Pp = P + (size_t)w * 8 * 32;
#pragma unroll
        for (int i = 0; i < 8; ++i) Pp[i * 32 + lane] = acc[i];
    }
    __syncthreads();

    // ---- reduce the 4 K-quarter partials and store bf16 context ----
    {
        const int i  = threadIdx.x >> 5;           // 0..7
        const int l2 = threadIdx.x & 31;
#pragma unroll
        for (int t2 = 0; t2 < 2; ++t2) {
            float s = 0.f;
#pragma unroll
            for (int kp = 0; kp < 4; ++kp) s += P[(((size_t)t2 * 4 + kp) * 8 + i) * 32 + l2];
            int srow = (l2 >> 4) * 8 + i;          // row within strip, 0..15
            int n = rt * 16 + srow;
            int d = t2 * 16 + (l2 & 15);
            s *= invRow[srow];
            ctx[((size_t)(b * NN + n)) * ED + h * HD + d] = (__bf16)s;
        }
    }
}

// ---------- kernel 3: output = ctx @ Woᵀ + bo ----------
__global__ __launch_bounds__(32)
void out_proj_kernel(const __bf16* __restrict__ ctx, const float* __restrict__ Wo,
                     const float* __restrict__ bo, float* __restrict__ out) {
    const int rt = blockIdx.x;                   // 0..255
    const int ct = blockIdx.y;                   // 0..15
    const int lane = threadIdx.x, half = lane >> 4, l = lane & 15;

    const __bf16* Ap = ctx + (size_t)(rt * 16 + l) * ED;
    const float*  Bp = Wo + (size_t)(ct * 16 + l) * ED;
    v8f acc = {};
#pragma unroll
    for (int k0 = 0; k0 < ED; k0 += 32) {
        v16bf a  = load_a_bf16(Ap + k0, half);
        v16bf bb = load_b_f32(Bp + k0, half);
        acc = wmma_bf16(a, bb, acc);
    }
#pragma unroll
    for (int i = 0; i < 8; ++i) {
        int row = rt * 16 + half * 8 + i;
        int col = ct * 16 + l;
        out[(size_t)row * ED + col] = acc[i] + bo[col];
    }
}

// ---------- host launcher ----------
extern "C" void kernel_launch(void* const* d_in, const int* in_sizes, int n_in,
                              void* d_out, int out_size, void* d_ws, size_t ws_size,
                              hipStream_t stream) {
    const float* features  = (const float*)d_in[0];
    const float* positions = (const float*)d_in[1];
    const unsigned char* mask = (const unsigned char*)d_in[2];
    const float* Wq = (const float*)d_in[3];  const float* bq = (const float*)d_in[4];
    const float* Wk = (const float*)d_in[5];  const float* bk = (const float*)d_in[6];
    const float* Wv = (const float*)d_in[7];  const float* bv = (const float*)d_in[8];
    const float* Wo = (const float*)d_in[9];  const float* bo = (const float*)d_in[10];
    const float* log_alpha = (const float*)d_in[11];
    const float* beta_     = (const float*)d_in[12];

    float* out  = (float*)d_out;                       // output (B,N,256) first
    float* attn = out + (size_t)BB * NN * ED;          // then attn (B,H,N,N)

    const size_t QKV = (size_t)BB * NH * NN * HD;      // 1,048,576 bf16 each
    __bf16* Qb  = (__bf16*)d_ws;
    __bf16* Kb  = Qb + QKV;
    __bf16* Vt  = Kb + QKV;
    __bf16* ctx = Vt + QKV;                            // total 8 MiB of ws

    proj_qkv_kernel<<<dim3(BN / 16, ED / 16, 3), 32, 0, stream>>>(
        features, Wq, bq, Wk, bk, Wv, bv, Qb, Kb, Vt);

    fused_attn_kernel<<<dim3(NN / 16, BB * NH), 256, SMEM_BYTES, stream>>>(
        Qb, Kb, Vt, positions, mask, log_alpha, beta_, attn, ctx);

    out_proj_kernel<<<dim3(BN / 16, ED / 16), 32, 0, stream>>>(ctx, Wo, bo, out);
}
